// SpatialReasoningGNN_40759239639285
// MI455X (gfx1250) — matrix operand order repaired and checked by reading, not compiled, not run on hardware
//
#include <hip/hip_runtime.h>
#include <hip/hip_bf16.h>
#include <stdint.h>

#define DCH 256      // hidden dim
#define NLAYERS 6
#define TILE 32      // rows (edges or nodes) per block

typedef __bf16 v16bf __attribute__((ext_vector_type(16)));
typedef float  v8f   __attribute__((ext_vector_type(8)));

union BFrag { uint4 u[2]; v16bf v; };

__device__ __forceinline__ unsigned short f2bf(float f) {
  unsigned int u = __float_as_uint(f);
  unsigned int r = (u + 0x7FFFu + ((u >> 16) & 1u)) >> 16;
  return (unsigned short)r;
}

__device__ __forceinline__ v16bf ldsA(const unsigned short* base, int row, int kb, int hi) {
  BFrag a;
  const uint4* ap = (const uint4*)&base[row * DCH + kb + hi * 16];
  a.u[0] = ap[0];
  a.u[1] = ap[1];           // A order used here (same as B): 16 contiguous K values k0..k0+15
  return a.v;
}

__device__ __forceinline__ v16bf gloB(const unsigned short* W, int col, int kb, int hi) {
  BFrag b;
  const uint4* bp = (const uint4*)&W[(long)col * DCH + kb + hi * 16];
  b.u[0] = bp[0];
  b.u[1] = bp[1];           // B layout: lane holds column, 16 contiguous K values
  return b.v;
}

// ---------------------------------------------------------------------------
// Weight convert + transpose: dst[l][n][k] = bf16(src[l*sls + (rowOff+k)*D + n])
// ---------------------------------------------------------------------------
__global__ void wconv_kernel(const float* __restrict__ src,
                             unsigned short* __restrict__ dst,
                             int layers, int rowOff, int srcLayerStride) {
  long total = (long)layers * DCH * DCH;
  for (long i = (long)blockIdx.x * blockDim.x + threadIdx.x; i < total;
       i += (long)gridDim.x * blockDim.x) {
    int k = (int)(i % DCH);
    int n = (int)((i / DCH) % DCH);
    int l = (int)(i / ((long)DCH * DCH));
    dst[(long)l * DCH * DCH + (long)n * DCH + k] =
        f2bf(src[(long)l * srcLayerStride + (long)(rowOff + k) * DCH + n]);
  }
}

// ---------------------------------------------------------------------------
// Node encoder: h = grid * node_W + node_b   (node_W is [1,D])
// ---------------------------------------------------------------------------
__global__ void encode_kernel(const int* __restrict__ grid,
                              const float* __restrict__ nodeW,
                              const float* __restrict__ nodeB,
                              float* __restrict__ h,
                              unsigned short* __restrict__ hb, long total) {
  for (long i = (long)blockIdx.x * blockDim.x + threadIdx.x; i < total;
       i += (long)gridDim.x * blockDim.x) {
    int n = (int)(i / DCH), d = (int)(i % DCH);
    float v = (float)grid[n] * nodeW[d] + nodeB[d];
    h[i] = v;
    hb[i] = f2bf(v);
  }
}

__global__ void zero_kernel(float* __restrict__ p, long total) {
  for (long i = (long)blockIdx.x * blockDim.x + threadIdx.x; i < total;
       i += (long)gridDim.x * blockDim.x) p[i] = 0.0f;
}

// ---------------------------------------------------------------------------
// Message kernel: per 32-edge tile
//   m1 = relu(h[src]@W1s + h[tgt]@W1t + ef@W1e + b1)   (WMMA + VALU edge term)
//   m  = m1@W2 + b2 ;  agg[tgt] += m  (f32 global atomics)
// Each wave register-blocks 4 column tiles (4 accumulators) so every A
// fragment read from LDS feeds 4 WMMAs.
// ---------------------------------------------------------------------------
__global__ __launch_bounds__(256)
void msg_kernel(const unsigned short* __restrict__ hb,
                const int* __restrict__ src, const int* __restrict__ tgt,
                const float* __restrict__ ef,
                const unsigned short* __restrict__ W1s,
                const unsigned short* __restrict__ W1t,
                const float* __restrict__ W1e, const float* __restrict__ b1,
                const unsigned short* __restrict__ W2,
                const float* __restrict__ b2,
                float* __restrict__ agg, int E) {
  __shared__ __align__(16) unsigned short As[TILE * DCH];
  __shared__ __align__(16) unsigned short At[TILE * DCH];
  __shared__ __align__(16) unsigned short M1[TILE * DCH];
  __shared__ float efs[TILE * 4];
  __shared__ int six[TILE];
  __shared__ int tix[TILE];

  const int tid = threadIdx.x;
  const int ebase = blockIdx.x * TILE;

  if (tid < TILE) {
    int e = min(ebase + tid, E - 1);
    six[tid] = src[e];
  } else if (tid < 2 * TILE) {
    int t2 = tid - TILE;
    int e = min(ebase + t2, E - 1);
    tix[t2] = tgt[e];
  }
  if (tid < TILE * 4) {
    int rr = tid >> 2, cc = tid & 3;
    int e = min(ebase + rr, E - 1);
    efs[tid] = ef[(long)e * 4 + cc];
  }
  __syncthreads();

  // Gather h[src] / h[tgt] rows (bf16, 512 B each) into LDS with 16B vectors.
  {
    int row = tid >> 3;           // 8 threads per row
    int ch  = (tid & 7) * 4;      // 4 x uint4 per thread
    const uint4* gs = (const uint4*)(hb + (long)six[row] * DCH);
    const uint4* gt = (const uint4*)(hb + (long)tix[row] * DCH);
    uint4* ls = (uint4*)&As[row * DCH];
    uint4* lt = (uint4*)&At[row * DCH];
#pragma unroll
    for (int j = 0; j < 4; ++j) { ls[ch + j] = gs[ch + j]; lt[ch + j] = gt[ch + j]; }
  }

  // Warm L2/L0 for stage-2 weights while stage 1 computes.
  {
    const char* p = (const char*)W2 + (size_t)tid * 512;
#pragma unroll
    for (int j = 0; j < 4; ++j) __builtin_prefetch(p + j * 128, 0, 3);
  }
  __syncthreads();

  const int lane = tid & 31;
  const int wave = tid >> 5;
  const int r  = wave & 1;        // 2 row tiles of 16
  const int cg = wave >> 1;       // 4 column groups, 4 tiles each
  const int m  = lane & 15;
  const int hi = lane >> 4;
  const int arow = r * 16 + m;

  // ---- Stage 1: hidden = relu(As@W1s + At@W1t + ef@W1e + b1) -> M1 (bf16)
  {
    v8f acc[4] = {};
#pragma unroll
    for (int kb = 0; kb < DCH; kb += 32) {
      v16bf a = ldsA(As, arow, kb, hi);
#pragma unroll
      for (int it = 0; it < 4; ++it) {
        v16bf b = gloB(W1s, (cg * 4 + it) * 16 + m, kb, hi);
        acc[it] = __builtin_amdgcn_wmma_f32_16x16x32_bf16(false, a, false, b,
                                                          (short)0, acc[it], false, false);
      }
      a = ldsA(At, arow, kb, hi);
#pragma unroll
      for (int it = 0; it < 4; ++it) {
        v16bf b = gloB(W1t, (cg * 4 + it) * 16 + m, kb, hi);
        acc[it] = __builtin_amdgcn_wmma_f32_16x16x32_bf16(false, a, false, b,
                                                          (short)0, acc[it], false, false);
      }
    }
#pragma unroll
    for (int it = 0; it < 4; ++it) {
      int c = cg * 4 + it;
#pragma unroll
      for (int i8 = 0; i8 < 8; ++i8) {
        int row = r * 16 + i8 + hi * 8;
        int col = c * 16 + m;
        float v = acc[it][i8] + b1[col];
#pragma unroll
        for (int j = 0; j < 4; ++j) v += efs[row * 4 + j] * W1e[j * DCH + col];
        v = fmaxf(v, 0.0f);
        M1[row * DCH + col] = f2bf(v);
      }
    }
  }
  __syncthreads();

  // ---- Stage 2: m = M1@W2 + b2 ; scatter-add into agg[tgt]
  {
    v8f acc[4] = {};
#pragma unroll
    for (int kb = 0; kb < DCH; kb += 32) {
      v16bf a = ldsA(M1, arow, kb, hi);
#pragma unroll
      for (int it = 0; it < 4; ++it) {
        v16bf b = gloB(W2, (cg * 4 + it) * 16 + m, kb, hi);
        acc[it] = __builtin_amdgcn_wmma_f32_16x16x32_bf16(false, a, false, b,
                                                          (short)0, acc[it], false, false);
      }
    }
#pragma unroll
    for (int it = 0; it < 4; ++it) {
      int c = cg * 4 + it;
#pragma unroll
      for (int i8 = 0; i8 < 8; ++i8) {
        int row = r * 16 + i8 + hi * 8;
        int col = c * 16 + m;
        if (ebase + row < E) {
          float v = acc[it][i8] + b2[col];
          atomicAdd(&agg[(long)tix[row] * DCH + col], v);
        }
      }
    }
  }
}

// ---------------------------------------------------------------------------
// Update kernel: per 32-node tile
//   u1 = relu(h@Ua + agg@Ub + b1) ; h_new = u1@U2 + b2
// ---------------------------------------------------------------------------
__global__ __launch_bounds__(256)
void upd_kernel(const float* __restrict__ agg,
                const unsigned short* __restrict__ Wa,
                const unsigned short* __restrict__ Wb,
                const float* __restrict__ b1,
                const unsigned short* __restrict__ W2,
                const float* __restrict__ b2,
                float* __restrict__ h, unsigned short* __restrict__ hb) {
  __shared__ __align__(16) unsigned short Ah[TILE * DCH];
  __shared__ __align__(16) unsigned short Ag[TILE * DCH];
  __shared__ __align__(16) unsigned short U1[TILE * DCH];

  const int tid = threadIdx.x;
  const int nbase = blockIdx.x * TILE;

  {
    int row = tid >> 3;
    int ch  = (tid & 7) * 4;
    const uint4* gh = (const uint4*)(hb + (long)(nbase + row) * DCH);
    uint4* lh = (uint4*)&Ah[row * DCH];
#pragma unroll
    for (int j = 0; j < 4; ++j) lh[ch + j] = gh[ch + j];

    const float4* ga = (const float4*)(agg + (long)(nbase + row) * DCH);
    int f4base = (tid & 7) * 8;
#pragma unroll
    for (int j = 0; j < 8; ++j) {
      float4 f = ga[f4base + j];
      unsigned int lo   = (unsigned int)f2bf(f.x) | ((unsigned int)f2bf(f.y) << 16);
      unsigned int hi32 = (unsigned int)f2bf(f.z) | ((unsigned int)f2bf(f.w) << 16);
      uint2 pk; pk.x = lo; pk.y = hi32;
      *(uint2*)&Ag[row * DCH + (f4base + j) * 4] = pk;
    }
  }
  {
    const char* p = (const char*)W2 + (size_t)tid * 512;
#pragma unroll
    for (int j = 0; j < 4; ++j) __builtin_prefetch(p + j * 128, 0, 3);
  }
  __syncthreads();

  const int lane = tid & 31;
  const int wave = tid >> 5;
  const int r  = wave & 1;
  const int cg = wave >> 1;
  const int m  = lane & 15;
  const int hi = lane >> 4;
  const int arow = r * 16 + m;

  // ---- Stage 1: U1 = relu(Ah@Wa + Ag@Wb + b1)
  {
    v8f acc[4] = {};
#pragma unroll
    for (int kb = 0; kb < DCH; kb += 32) {
      v16bf a = ldsA(Ah, arow, kb, hi);
#pragma unroll
      for (int it = 0; it < 4; ++it) {
        v16bf b = gloB(Wa, (cg * 4 + it) * 16 + m, kb, hi);
        acc[it] = __builtin_amdgcn_wmma_f32_16x16x32_bf16(false, a, false, b,
                                                          (short)0, acc[it], false, false);
      }
      a = ldsA(Ag, arow, kb, hi);
#pragma unroll
      for (int it = 0; it < 4; ++it) {
        v16bf b = gloB(Wb, (cg * 4 + it) * 16 + m, kb, hi);
        acc[it] = __builtin_amdgcn_wmma_f32_16x16x32_bf16(false, a, false, b,
                                                          (short)0, acc[it], false, false);
      }
    }
#pragma unroll
    for (int it = 0; it < 4; ++it) {
      int c = cg * 4 + it;
#pragma unroll
      for (int i8 = 0; i8 < 8; ++i8) {
        int row = r * 16 + i8 + hi * 8;
        int col = c * 16 + m;
        float v = fmaxf(acc[it][i8] + b1[col], 0.0f);
        U1[row * DCH + col] = f2bf(v);
      }
    }
  }
  __syncthreads();

  // ---- Stage 2: h_new = U1@W2 + b2
  {
    v8f acc[4] = {};
#pragma unroll
    for (int kb = 0; kb < DCH; kb += 32) {
      v16bf a = ldsA(U1, arow, kb, hi);
#pragma unroll
      for (int it = 0; it < 4; ++it) {
        v16bf b = gloB(W2, (cg * 4 + it) * 16 + m, kb, hi);
        acc[it] = __builtin_amdgcn_wmma_f32_16x16x32_bf16(false, a, false, b,
                                                          (short)0, acc[it], false, false);
      }
    }
#pragma unroll
    for (int it = 0; it < 4; ++it) {
      int c = cg * 4 + it;
#pragma unroll
      for (int i8 = 0; i8 < 8; ++i8) {
        int row = r * 16 + i8 + hi * 8;
        int col = c * 16 + m;
        float v = acc[it][i8] + b2[col];
        long gi = (long)(nbase + row) * DCH + col;
        h[gi] = v;
        hb[gi] = f2bf(v);
      }
    }
  }
}

// ---------------------------------------------------------------------------
// Output head: out[n][c] = h[n] . out_W[:,c] + out_b[c]   (D=256 -> 10)
// ---------------------------------------------------------------------------
__global__ void out_kernel(const float* __restrict__ h,
                           const float* __restrict__ Wo,
                           const float* __restrict__ bo,
                           float* __restrict__ out, long total) {
  for (long i = (long)blockIdx.x * blockDim.x + threadIdx.x; i < total;
       i += (long)gridDim.x * blockDim.x) {
    int n = (int)(i / 10), c = (int)(i % 10);
    float acc = bo[c];
    const float* hr = h + (long)n * DCH;
#pragma unroll 8
    for (int d = 0; d < DCH; ++d) acc += hr[d] * Wo[d * 10 + c];
    out[i] = acc;
  }
}

// ---------------------------------------------------------------------------
extern "C" void kernel_launch(void* const* d_in, const int* in_sizes, int n_in,
                              void* d_out, int out_size, void* d_ws, size_t ws_size,
                              hipStream_t stream) {
  const int*   grid  = (const int*)d_in[0];
  const int*   eidx  = (const int*)d_in[1];
  const float* ef    = (const float*)d_in[2];
  const float* nodeW = (const float*)d_in[3];
  const float* nodeB = (const float*)d_in[4];
  const float* mW1   = (const float*)d_in[5];
  const float* mb1   = (const float*)d_in[6];
  const float* mW2   = (const float*)d_in[7];
  const float* mb2   = (const float*)d_in[8];
  const float* uW1   = (const float*)d_in[9];
  const float* ub1   = (const float*)d_in[10];
  const float* uW2   = (const float*)d_in[11];
  const float* ub2   = (const float*)d_in[12];
  const float* oW    = (const float*)d_in[13];
  const float* ob    = (const float*)d_in[14];

  const int N = in_sizes[0];          // 65536 nodes
  const int E = in_sizes[1] / 2;      // edges
  const int* srcIdx = eidx;
  const int* tgtIdx = eidx + E;

  // Workspace layout
  char* ws = (char*)d_ws;
  size_t off = 0;
  unsigned short* hb  = (unsigned short*)(ws + off); off += (size_t)N * DCH * 2;
  float* h   = (float*)(ws + off); off += (size_t)N * DCH * 4;
  float* agg = (float*)(ws + off); off += (size_t)N * DCH * 4;
  const size_t WSZ = (size_t)NLAYERS * DCH * DCH;   // elements per weight set
  unsigned short* W1s = (unsigned short*)(ws + off); off += WSZ * 2;
  unsigned short* W1t = (unsigned short*)(ws + off); off += WSZ * 2;
  unsigned short* W2m = (unsigned short*)(ws + off); off += WSZ * 2;
  unsigned short* Ua  = (unsigned short*)(ws + off); off += WSZ * 2;
  unsigned short* Ub  = (unsigned short*)(ws + off); off += WSZ * 2;
  unsigned short* U2  = (unsigned short*)(ws + off); off += WSZ * 2;

  // 1) Convert + transpose weights to bf16 [n][k] layout
  const int cvtBlocks = 1536;
  wconv_kernel<<<cvtBlocks, 256, 0, stream>>>(mW1, W1s, NLAYERS,   0, 516 * DCH);
  wconv_kernel<<<cvtBlocks, 256, 0, stream>>>(mW1, W1t, NLAYERS, 256, 516 * DCH);
  wconv_kernel<<<cvtBlocks, 256, 0, stream>>>(mW2, W2m, NLAYERS,   0, DCH * DCH);
  wconv_kernel<<<cvtBlocks, 256, 0, stream>>>(uW1, Ua,  NLAYERS,   0, 512 * DCH);
  wconv_kernel<<<cvtBlocks, 256, 0, stream>>>(uW1, Ub,  NLAYERS, 256, 512 * DCH);
  wconv_kernel<<<cvtBlocks, 256, 0, stream>>>(uW2, U2,  NLAYERS,   0, DCH * DCH);

  // 2) Node encoder
  encode_kernel<<<4096, 256, 0, stream>>>(grid, nodeW, nodeB, h, hb, (long)N * DCH);

  // 3) Message-passing layers
  const int msgBlocks = (E + TILE - 1) / TILE;
  const int updBlocks = N / TILE;
  for (int l = 0; l < NLAYERS; ++l) {
    zero_kernel<<<4096, 256, 0, stream>>>(agg, (long)N * DCH);
    msg_kernel<<<msgBlocks, 256, 0, stream>>>(
        hb, srcIdx, tgtIdx, ef,
        W1s + (size_t)l * DCH * DCH, W1t + (size_t)l * DCH * DCH,
        mW1 + (size_t)l * 516 * DCH + 512 * DCH, mb1 + (size_t)l * DCH,
        W2m + (size_t)l * DCH * DCH, mb2 + (size_t)l * DCH,
        agg, E);
    upd_kernel<<<updBlocks, 256, 0, stream>>>(
        agg,
        Ua + (size_t)l * DCH * DCH, Ub + (size_t)l * DCH * DCH,
        ub1 + (size_t)l * DCH,
        U2 + (size_t)l * DCH * DCH, ub2 + (size_t)l * DCH,
        h, hb);
  }

  // 4) Output head
  out_kernel<<<2048, 256, 0, stream>>>(h, oW, ob, (float*)d_out, (long)N * 10);
}
